// HSMM_82626580840586
// MI455X (gfx1250) — matrix operand order repaired and checked by reading, not compiled, not run on hardware
//
#include <hip/hip_runtime.h>
#include <hip/hip_bf16.h>

typedef __attribute__((ext_vector_type(16))) _Float16 v16h;
typedef __attribute__((ext_vector_type(8)))  _Float16 v8h;
typedef __attribute__((ext_vector_type(8)))  float    v8f;

#define KST  5
#define LSEG 4
#define EMBD 300
#define SED  600
#define HID  512
#define RIN  900
#define GTN  10000
#define BSZ  8
#define TLEN 48
#define NN   (BSZ*TLEN)   /* 384 */
#define EPAD 320
#define G4   2048         /* 4*HID */
#define NDEC 10048        /* GT+1 padded to multiple of 64 */
#define DSTR 10016
#define NEGV (-1e30f)

__device__ __forceinline__ float sigf(float x) { return 1.f / (1.f + __expf(-x)); }

// ---------------------------------------------------------------------------
// Generic WMMA GEMM: C[M,N] = A[M,K] * W[N,K]^T   (A,W f16 row-major, C f32)
// One wave per block; wave computes a 16x64 tile (4 wmma accumulators share A).
// M%16==0, N%64==0, K%32==0; all rows 16B aligned.
// ---------------------------------------------------------------------------
__global__ void __launch_bounds__(32)
k_gemm16(const _Float16* __restrict__ A, const _Float16* __restrict__ W,
         float* __restrict__ C, int M, int N, int K) {
  const int m0   = blockIdx.x << 4;
  const int n0   = blockIdx.y << 6;
  const int lane = threadIdx.x & 31;
  const int half = lane >> 4;
  const int l16  = lane & 15;

  v8f z = {};
  v8f acc0 = z, acc1 = z, acc2 = z, acc3 = z;

  const _Float16* ap  = A + (size_t)(m0 + l16) * K + half * 8;   // A frag base
  const _Float16* wp0 = W + (size_t)(n0 + l16) * K + half * 8;   // B frag base

  for (int kk = 0; kk < K; kk += 32) {
    // A fragment (16x32): lane<16 holds K kk+0..7 / kk+16..23, lane>=16 kk+8..15 / kk+24..31
    v16h af;
    {
      v8h lo = *(const v8h*)(ap + kk);
      v8h hi = *(const v8h*)(ap + kk + 16);
#pragma unroll
      for (int i = 0; i < 8; ++i) { af[i] = lo[i]; af[i + 8] = hi[i]; }
    }
#pragma unroll
    for (int j = 0; j < 4; ++j) {
      // B fragment (32x16): lane = column n, lane<16 holds K 0..7 / 16..23, lane>=16 8..15 / 24..31
      const _Float16* wp = wp0 + (size_t)(j * 16) * K + kk;
      v8h lo = *(const v8h*)(wp);
      v8h hi = *(const v8h*)(wp + 16);
      v16h bf;
#pragma unroll
      for (int i = 0; i < 8; ++i) { bf[i] = lo[i]; bf[i + 8] = hi[i]; }
      v8f a = (j == 0) ? acc0 : (j == 1) ? acc1 : (j == 2) ? acc2 : acc3;
      a = __builtin_amdgcn_wmma_f32_16x16x32_f16(false, af, false, bf,
                                                 (short)0, a, false, false);
      if (j == 0) acc0 = a; else if (j == 1) acc1 = a; else if (j == 2) acc2 = a; else acc3 = a;
    }
  }
  // C/D layout: lane = N (within tile), vgpr r = M row (lanes>=16 -> M=8+r)
#pragma unroll
  for (int j = 0; j < 4; ++j) {
    v8f a = (j == 0) ? acc0 : (j == 1) ? acc1 : (j == 2) ? acc2 : acc3;
    float* cp = C + (size_t)(m0 + half * 8) * N + n0 + j * 16 + l16;
#pragma unroll
    for (int r = 0; r < 8; ++r) cp[(size_t)r * N] = a[r];
  }
}

// ---------------------------------------------------------------------------
// f32 -> f16 pack with row/col zero padding and source column offset
// ---------------------------------------------------------------------------
__global__ void k_pack16(const float* __restrict__ src, _Float16* __restrict__ dst,
                         int Rdst, int Cdst, int Rvalid, int Ccopy, int srcLd, int srcOff) {
  int i = blockIdx.x * blockDim.x + threadIdx.x;
  if (i >= Rdst * Cdst) return;
  int r = i / Cdst, c = i % Cdst;
  float v = (r < Rvalid && c < Ccopy) ? src[(size_t)r * srcLd + srcOff + c] : 0.f;
  dst[i] = (_Float16)v;
}

// Build segment embeddings [L+1][N][EPAD] (f16, padded 300->320)
__global__ void k_segembs(const int* __restrict__ x, const float* __restrict__ lut,
                          const float* __restrict__ start_emb, _Float16* __restrict__ seg) {
  int i = blockIdx.x * blockDim.x + threadIdx.x;
  if (i >= (LSEG + 1) * NN * EPAD) return;
  int e = i % EPAD;
  int n = (i / EPAD) % NN;
  int l = i / (EPAD * NN);
  float v = 0.f;
  if (e < EMBD) {
    if (l == 0) v = start_emb[e];
    else {
      int b = n / TLEN, tt = n % TLEN;
      int tp = tt + (l - 1);
      if (tp < TLEN) { int w = x[tp * BSZ + b]; v = lut[(size_t)w * EMBD + e]; }
    }
  }
  seg[i] = (_Float16)v;
}

// Per-state LSTM gate bias: bih + bhh + Wih[:,300:900] @ se_k
__global__ void k_biasK(const float* __restrict__ Wih, const float* __restrict__ bih,
                        const float* __restrict__ bhh, const float* __restrict__ se,
                        float* __restrict__ biasK) {
  int i = blockIdx.x * blockDim.x + threadIdx.x;
  if (i >= KST * G4) return;
  int j = i % G4, k = i / G4;
  const float* wr = Wih + (size_t)j * RIN + EMBD;
  const float* sk = se + k * SED;
  float s = bih[j] + bhh[j];
  for (int q = 0; q < SED; ++q) s += wr[q] * sk[q];
  biasK[i] = s;
}

// Per-state category distribution (tiny MLP + softmax), one block of HID threads
__global__ void k_cat(const float* __restrict__ W1, const float* __restrict__ b1,
                      const float* __restrict__ W2, const float* __restrict__ b2,
                      const float* __restrict__ se, float* __restrict__ cat_sm) {
  __shared__ float h1[HID];
  __shared__ float lg[3];
  int tid = threadIdx.x;
  for (int k = 0; k < KST; ++k) {
    const float* sk = se + k * SED;
    float a = b1[tid];
    const float* wr = W1 + (size_t)tid * SED;
    for (int q = 0; q < SED; ++q) a += wr[q] * sk[q];
    h1[tid] = a > 0.f ? a : 0.f;
    __syncthreads();
    if (tid < 3) {
      float c = b2[tid];
      const float* wr2 = W2 + (size_t)tid * HID;
      for (int h = 0; h < HID; ++h) c += wr2[h] * h1[h];
      lg[tid] = c;
    }
    __syncthreads();
    if (tid == 0) {
      float m = fmaxf(lg[0], fmaxf(lg[1], lg[2]));
      float e0 = __expf(lg[0] - m), e1 = __expf(lg[1] - m), e2 = __expf(lg[2] - m);
      float s = e0 + e1 + e2;
      cat_sm[k * 3 + 0] = e0 / s; cat_sm[k * 3 + 1] = e1 / s; cat_sm[k * 3 + 2] = e2 / s;
    }
    __syncthreads();
  }
}

// log(vocab mask) per state: log(sum_c cat[c]*vm[c][j]); j==GT -> 0
__global__ void k_logvm(const float* __restrict__ cat_sm, const float* __restrict__ vm,
                        float* __restrict__ logvm) {
  int i = blockIdx.x * blockDim.x + threadIdx.x;
  if (i >= KST * DSTR) return;
  int j = i % DSTR, k = i / DSTR;
  float v = 0.f;
  if (j < GTN) {
    v = logf(cat_sm[k * 3 + 0] * vm[j] + cat_sm[k * 3 + 1] * vm[GTN + j] +
             cat_sm[k * 3 + 2] * vm[2 * GTN + j]);
  }
  logvm[i] = v;
}

// Per-state decoder bias: decb + decW @ out_biases_k
__global__ void k_decbK(const float* __restrict__ decW, const float* __restrict__ decb,
                        const float* __restrict__ out_biases, float* __restrict__ decbK) {
  int i = blockIdx.x * blockDim.x + threadIdx.x;
  if (i >= KST * DSTR) return;
  int j = i % DSTR, k = i / DSTR;
  float s = 0.f;
  if (j <= GTN) {
    s = decb[j];
    const float* wr = decW + (size_t)j * HID;
    const float* ob = out_biases + k * HID;
    for (int h = 0; h < HID; ++h) s += wr[h] * ob[h];
  }
  decbK[i] = s;
}

// LSTM state init: h = tanh(h0_lin[:H]) (f16), c = h0_lin[H:]
__global__ void k_init(const float* __restrict__ h0_lin, _Float16* __restrict__ h16,
                       float* __restrict__ c) {
  int i = blockIdx.x * blockDim.x + threadIdx.x;
  if (i >= NN * HID) return;
  int j = i % HID;
  h16[i] = (_Float16)tanhf(h0_lin[j]);
  c[i] = h0_lin[HID + j];
}

// LSTM pointwise cell; writes h (f16) and scaled h (for the decoder GEMM)
__global__ void k_lstm(const float* __restrict__ gatesH, const float* __restrict__ xe_t,
                       const float* __restrict__ bK, const float* __restrict__ og,
                       _Float16* __restrict__ h16, float* __restrict__ c,
                       _Float16* __restrict__ hs_t) {
  int i = blockIdx.x * blockDim.x + threadIdx.x;
  if (i >= NN * HID) return;
  int j = i % HID, n = i / HID;
  size_t base = (size_t)n * G4;
  float gi = gatesH[base + j]           + xe_t[base + j]           + bK[j];
  float gf = gatesH[base + HID + j]     + xe_t[base + HID + j]     + bK[HID + j];
  float gg = gatesH[base + 2 * HID + j] + xe_t[base + 2 * HID + j] + bK[2 * HID + j];
  float go = gatesH[base + 3 * HID + j] + xe_t[base + 3 * HID + j] + bK[3 * HID + j];
  float cc = sigf(gf) * c[i] + sigf(gi) * tanhf(gg);
  float h = sigf(go) * tanhf(cc);
  c[i] = cc;
  h16[i] = (_Float16)h;
  hs_t[i] = (_Float16)(h * og[j]);
}

// Fused masked log-softmax row reduction (online LSE), one block per row
__global__ void k_reduce(const float* __restrict__ logits, const float* __restrict__ decb_k,
                         const float* __restrict__ logvm_k, const int* __restrict__ combotargs,
                         int t, float* __restrict__ lls_k, float* __restrict__ eop_k) {
  __shared__ float sm[256], ss[256];
  int n = blockIdx.x, tid = threadIdx.x;
  const float* row = logits + (size_t)n * NDEC;
  float m = -3.0e38f, s = 0.f;
  for (int j = tid; j <= GTN; j += 256) {
    float v = row[j] + decb_k[j] + logvm_k[j];
    if (v > m) { s = s * __expf(m - v) + 1.f; m = v; }
    else       { s += __expf(v - m); }
  }
  sm[tid] = m; ss[tid] = s;
  __syncthreads();
  for (int st = 128; st > 0; st >>= 1) {
    if (tid < st) {
      float m2 = sm[tid + st], s2 = ss[tid + st];
      if (m2 > sm[tid]) { ss[tid] = ss[tid] * __expf(sm[tid] - m2) + s2; sm[tid] = m2; }
      else              { ss[tid] += s2 * __expf(m2 - sm[tid]); }
    }
    __syncthreads();
  }
  if (tid == 0) {
    float lse = sm[0] + logf(ss[0]);
    if (t < LSEG) {
      int tgt = combotargs[t * NN + n];
      float vt = row[tgt] + decb_k[tgt] + logvm_k[tgt];
      lls_k[t * NN + n] = vt - lse;
    }
    if (t >= 1) {
      float ve = row[GTN] + decb_k[GTN];  // logvm at EOP == 0
      eop_k[(t - 1) * NN + n] = ve - lse;
    }
  }
}

// Transition/length log-softmaxes + semi-Markov log-space scan; one block
__global__ void k_trellis(const float* __restrict__ lls, const float* __restrict__ eop,
                          const float* __restrict__ se, const float* __restrict__ tw,
                          const float* __restrict__ tb, const float* __restrict__ init_trans,
                          const float* __restrict__ lenW, const float* __restrict__ lenb,
                          float* __restrict__ out) {
  __shared__ float u[KST * SED];
  __shared__ float trans[KST * KST];
  __shared__ float initl[KST];
  __shared__ float ls[KST * LSEG];
  __shared__ float lenlp[TLEN * LSEG * KST];
  __shared__ float obs[LSEG * TLEN * BSZ * KST];
  __shared__ float buf[LSEG * BSZ * KST];
  __shared__ float alpha[BSZ * KST];
  int tid = threadIdx.x;

  for (int i = tid; i < KST * SED; i += 256) {
    int k = i / SED, s = i % SED;
    float a = 0.f;
    for (int q = 0; q < SED; ++q) a += se[k * SED + q] * tw[q * SED + s];
    u[i] = a;
  }
  __syncthreads();
  if (tid < KST * KST) {
    int k = tid / KST, kk = tid % KST;
    float a = tb[tid];
    for (int s = 0; s < SED; ++s) a += u[k * SED + s] * se[kk * SED + s];
    trans[tid] = (k == kk) ? NEGV : a;
  }
  if (tid < KST * LSEG) {
    int k = tid / LSEG, l = tid % LSEG;
    float a = lenb[l];
    for (int s = 0; s < SED; ++s) a += se[k * SED + s] * lenW[l * SED + s];
    ls[tid] = a;
  }
  __syncthreads();
  if (tid < KST) {  // row-wise log-softmax of transitions
    float m = NEGV;
    for (int kk = 0; kk < KST; ++kk) m = fmaxf(m, trans[tid * KST + kk]);
    float s = 0.f;
    for (int kk = 0; kk < KST; ++kk) s += __expf(trans[tid * KST + kk] - m);
    float l = m + logf(s);
    for (int kk = 0; kk < KST; ++kk) trans[tid * KST + kk] -= l;
  }
  if (tid == 0) {  // init log-softmax
    float m = NEGV;
    for (int k = 0; k < KST; ++k) m = fmaxf(m, init_trans[k]);
    float s = 0.f;
    for (int k = 0; k < KST; ++k) s += __expf(init_trans[k] - m);
    float l = m + logf(s);
    for (int k = 0; k < KST; ++k) initl[k] = init_trans[k] - l;
  }
  __syncthreads();
  if (tid < TLEN * KST) {  // length log-softmax over feasible lengths
    int t = tid / KST, k = tid % KST;
    int rem = TLEN - t; if (rem > LSEG) rem = LSEG;
    float v[LSEG]; float m = NEGV;
    for (int l = 0; l < LSEG; ++l) { v[l] = (l < rem) ? ls[k * LSEG + l] : NEGV; m = fmaxf(m, v[l]); }
    float s = 0.f;
    for (int l = 0; l < LSEG; ++l) s += __expf(v[l] - m);
    float lse = m + logf(s);
    for (int l = 0; l < LSEG; ++l) lenlp[(t * LSEG + l) * KST + k] = v[l] - lse;
  }
  for (int i = tid; i < LSEG * TLEN * BSZ * KST; i += 256) {
    int k = i % KST, b = (i / KST) % BSZ, t = (i / (KST * BSZ)) % TLEN, l = i / (KST * BSZ * TLEN);
    int n = b * TLEN + t;
    float cum = 0.f;
    for (int j = 0; j <= l; ++j) cum += lls[(k * LSEG + j) * NN + n];
    obs[i] = cum + eop[(k * LSEG + l) * NN + n];
  }
  for (int i = tid; i < LSEG * BSZ * KST; i += 256) buf[i] = NEGV;
  __syncthreads();
  if (tid < BSZ * KST) buf[tid] = initl[tid % KST];
  __syncthreads();

  int head = 0;
  for (int t = 0; t < TLEN; ++t) {
    if (tid < BSZ * KST) {
      int b = tid / KST, k = tid % KST;
      float v[LSEG]; float m = NEGV;
      for (int l = 0; l < LSEG; ++l) {
        int phys = (head + l) & (LSEG - 1);
        float bv = buf[phys * (BSZ * KST) + tid];
        float o  = (t >= l) ? obs[((l * TLEN + (t - l)) * BSZ + b) * KST + k] : NEGV;
        float le = (t >= l) ? lenlp[((t - l) * LSEG + l) * KST + k] : NEGV;
        v[l] = bv + le + o;
        m = fmaxf(m, v[l]);
      }
      float s = 0.f;
      for (int l = 0; l < LSEG; ++l) s += __expf(v[l] - m);
      alpha[tid] = m + logf(s);
    }
    __syncthreads();
    int nh = (head + LSEG - 1) & (LSEG - 1);
    if (tid < BSZ * KST) {
      int b = tid / KST, k = tid % KST;
      float v[KST]; float m = NEGV;
      for (int kk = 0; kk < KST; ++kk) { v[kk] = alpha[b * KST + kk] + trans[kk * KST + k]; m = fmaxf(m, v[kk]); }
      float s = 0.f;
      for (int kk = 0; kk < KST; ++kk) s += __expf(v[kk] - m);
      buf[nh * (BSZ * KST) + tid] = m + logf(s);
    }
    __syncthreads();
    head = nh;
  }
  if (tid < BSZ) {
    float m = NEGV;
    for (int k = 0; k < KST; ++k) m = fmaxf(m, alpha[tid * KST + k]);
    float s = 0.f;
    for (int k = 0; k < KST; ++k) s += __expf(alpha[tid * KST + k] - m);
    out[tid] = m + logf(s);
  }
}

// ---------------------------------------------------------------------------
extern "C" void kernel_launch(void* const* d_in, const int* in_sizes, int n_in,
                              void* d_out, int out_size, void* d_ws, size_t ws_size,
                              hipStream_t stream) {
  const int*   x            = (const int*)d_in[0];
  const int*   combotargs   = (const int*)d_in[1];
  const float* vocab_masks  = (const float*)d_in[2];
  const float* lut          = (const float*)d_in[3];
  const float* start_emb    = (const float*)d_in[4];
  const float* state_embs   = (const float*)d_in[6];
  const float* trans_weights= (const float*)d_in[7];
  const float* trans_bias   = (const float*)d_in[8];
  const float* init_trans   = (const float*)d_in[9];
  const float* h0_lin       = (const float*)d_in[10];
  const float* Wih          = (const float*)d_in[11];
  const float* Whh          = (const float*)d_in[12];
  const float* bih          = (const float*)d_in[13];
  const float* bhh          = (const float*)d_in[14];
  const float* out_gates    = (const float*)d_in[15];
  const float* out_biases   = (const float*)d_in[16];
  const float* decW         = (const float*)d_in[17];
  const float* decb         = (const float*)d_in[18];
  const float* cat_W1       = (const float*)d_in[19];
  const float* cat_b1       = (const float*)d_in[20];
  const float* cat_W2       = (const float*)d_in[21];
  const float* cat_b2       = (const float*)d_in[22];
  const float* len_W        = (const float*)d_in[23];
  const float* len_b        = (const float*)d_in[24];
  float* out = (float*)d_out;

  size_t off = 0;
  auto alloc = [&](size_t bytes) -> void* {
    off = (off + 255) & ~(size_t)255;
    void* p = (char*)d_ws + off;
    off += bytes;
    return p;
  };
  _Float16* Whh16   = (_Float16*)alloc((size_t)G4 * HID * 2);
  _Float16* WihE16  = (_Float16*)alloc((size_t)G4 * EPAD * 2);
  _Float16* decW16  = (_Float16*)alloc((size_t)NDEC * HID * 2);
  _Float16* seg16   = (_Float16*)alloc((size_t)(LSEG + 1) * NN * EPAD * 2);
  float*    XE      = (float*)alloc((size_t)(LSEG + 1) * NN * G4 * 4);
  float*    biasK   = (float*)alloc((size_t)KST * G4 * 4);
  float*    decbK   = (float*)alloc((size_t)KST * DSTR * 4);
  float*    cat_sm  = (float*)alloc((size_t)KST * 3 * 4);
  float*    logvm   = (float*)alloc((size_t)KST * DSTR * 4);
  _Float16* h16     = (_Float16*)alloc((size_t)NN * HID * 2);
  float*    cbuf    = (float*)alloc((size_t)NN * HID * 4);
  float*    gatesH  = (float*)alloc((size_t)NN * G4 * 4);
  _Float16* hs16    = (_Float16*)alloc((size_t)(LSEG + 1) * NN * HID * 2);
  float*    logits  = (float*)alloc((size_t)NN * NDEC * 4);
  float*    lls     = (float*)alloc((size_t)KST * LSEG * NN * 4);
  float*    eopb    = (float*)alloc((size_t)KST * LSEG * NN * 4);

  const int tpb = 256;
  auto G = [&](long long n) { return (int)((n + tpb - 1) / tpb); };

  // Weight packing + state-independent precompute
  k_pack16<<<G((long long)G4 * HID), tpb, 0, stream>>>(Whh, Whh16, G4, HID, G4, HID, HID, 0);
  k_pack16<<<G((long long)G4 * EPAD), tpb, 0, stream>>>(Wih, WihE16, G4, EPAD, G4, EMBD, RIN, 0);
  k_pack16<<<G((long long)NDEC * HID), tpb, 0, stream>>>(decW, decW16, NDEC, HID, GTN + 1, HID, HID, 0);
  k_segembs<<<G((long long)(LSEG + 1) * NN * EPAD), tpb, 0, stream>>>(x, lut, start_emb, seg16);
  // XE[t] = segembs[t] @ WihE^T  (state independent; done once for all 5 steps)
  k_gemm16<<<dim3((LSEG + 1) * NN / 16, G4 / 64), 32, 0, stream>>>(seg16, WihE16, XE,
                                                                   (LSEG + 1) * NN, G4, EPAD);
  k_biasK<<<G((long long)KST * G4), tpb, 0, stream>>>(Wih, bih, bhh, state_embs, biasK);
  k_cat<<<1, HID, 0, stream>>>(cat_W1, cat_b1, cat_W2, cat_b2, state_embs, cat_sm);
  k_logvm<<<G((long long)KST * DSTR), tpb, 0, stream>>>(cat_sm, vocab_masks, logvm);
  k_decbK<<<G((long long)KST * DSTR), tpb, 0, stream>>>(decW, decb, out_biases, decbK);

  for (int k = 0; k < KST; ++k) {
    k_init<<<G((long long)NN * HID), tpb, 0, stream>>>(h0_lin, h16, cbuf);
    for (int t = 0; t <= LSEG; ++t) {
      k_gemm16<<<dim3(NN / 16, G4 / 64), 32, 0, stream>>>(h16, Whh16, gatesH, NN, G4, HID);
      k_lstm<<<G((long long)NN * HID), tpb, 0, stream>>>(
          gatesH, XE + (size_t)t * NN * G4, biasK + (size_t)k * G4, out_gates + (size_t)k * HID,
          h16, cbuf, hs16 + (size_t)t * NN * HID);
    }
    for (int t = 0; t <= LSEG; ++t) {
      k_gemm16<<<dim3(NN / 16, NDEC / 64), 32, 0, stream>>>(
          hs16 + (size_t)t * NN * HID, decW16, logits, NN, NDEC, HID);
      k_reduce<<<NN, 256, 0, stream>>>(logits, decbK + (size_t)k * DSTR,
                                       logvm + (size_t)k * DSTR, combotargs, t,
                                       lls + (size_t)k * LSEG * NN, eopb + (size_t)k * LSEG * NN);
    }
  }
  k_trellis<<<1, 256, 0, stream>>>(lls, eopb, state_embs, trans_weights, trans_bias,
                                   init_trans, len_W, len_b, out);
}
